// WorldModel_44152263802896
// MI455X (gfx1250) — matrix operand order repaired and checked by reading, not compiled, not run on hardware
//
#include <hip/hip_runtime.h>
#include <hip/hip_bf16.h>

// ---------------------------------------------------------------------------
// DreamerV3 world-model forward loss on MI455X (gfx1250, wave32, WMMA).
// GEMMs: v_wmma_f32_16x16x32_f16, f16 A staged into LDS via CDNA5 async
// copies (global_load_async_to_lds_b128, ASYNCcnt), f16 B (weights
// pre-transposed to [N][K]) streamed from L2, f32 accumulation.
// Block tile 32(M) x 256(N): each B fragment is reused for two WMMAs.
// ---------------------------------------------------------------------------

typedef __attribute__((ext_vector_type(16))) _Float16 v16h;
typedef __attribute__((ext_vector_type(8)))  _Float16 v8h;
typedef __attribute__((ext_vector_type(8)))  float    v8f;

#define BB    1024   // batch
#define LL    64     // sequence length
#define OBSD  256
#define ACTD  32
#define STO   64     // stoch
#define DET   1024   // deter
#define HIDD  512
#define FEATD 1088   // DET + STO
#define NB    255    // reward bins

// ------------------------------- reductions --------------------------------
__device__ __forceinline__ float waveReduceSum(float v) {
#pragma unroll
  for (int o = 16; o > 0; o >>= 1) v += __shfl_xor(v, o, 32);
  return v;
}
__device__ __forceinline__ float waveReduceMax(float v) {
#pragma unroll
  for (int o = 16; o > 0; o >>= 1) v = fmaxf(v, __shfl_xor(v, o, 32));
  return v;
}
__device__ __forceinline__ float blockReduceSumAll(float v) {
  __shared__ float sm[9];
  const int lane = threadIdx.x & 31, w = threadIdx.x >> 5, nw = blockDim.x >> 5;
  v = waveReduceSum(v);
  __syncthreads();
  if (lane == 0) sm[w] = v;
  __syncthreads();
  if (w == 0) {
    float x = (lane < nw) ? sm[lane] : 0.f;
    x = waveReduceSum(x);
    if (lane == 0) sm[8] = x;
  }
  __syncthreads();
  return sm[8];
}
__device__ __forceinline__ float blockReduceMaxAll(float v) {
  __shared__ float smx[9];
  const int lane = threadIdx.x & 31, w = threadIdx.x >> 5, nw = blockDim.x >> 5;
  v = waveReduceMax(v);
  __syncthreads();
  if (lane == 0) smx[w] = v;
  __syncthreads();
  if (w == 0) {
    float x = (lane < nw) ? smx[lane] : -3.0e38f;
    x = waveReduceMax(x);
    if (lane == 0) smx[8] = x;
  }
  __syncthreads();
  return smx[8];
}
__device__ __forceinline__ float sigmoidf_(float x) { return 1.f / (1.f + expf(-x)); }
__device__ __forceinline__ float softplusf_(float x) {
  return (x > 20.f) ? x : log1pf(expf(x));
}

// ------------------------------- WMMA GEMM ---------------------------------
// C[M,N] = act(A[M,K] * Wt^T + bias), Wt stored as [Npad][K] f16 (K-contig).
// Block: 128 threads = 4 waves. Block tile: 32(M) x 256(N); wave tile 32x64.
// A tile (32x32 f16, double buffered) is DMA'd into LDS with
// global_load_async_to_lds_b128; B fragments come straight from Wt (L2).
template <int ACTIV>
__global__ __launch_bounds__(128) void wmma_gemm(
    const _Float16* __restrict__ A, int lda,
    const _Float16* __restrict__ Wt,
    const float* __restrict__ bias,
    float* __restrict__ C32, int ldc32,
    _Float16* __restrict__ C16, int ldc16,
    int M, int N, int K) {
  __shared__ alignas(16) _Float16 sA[2][32][32];

  const int tid  = threadIdx.x;
  const int lane = tid & 31;
  const int wave = tid >> 5;
  const int g    = lane >> 4;   // half-wave group (0/1)
  const int mh   = lane & 15;   // A-row / B-col within a 16-tile
  const int mb   = blockIdx.y * 32;
  const int nb0  = blockIdx.x * 256 + wave * 64;

  // async staging: each of 128 threads DMAs one 16B chunk of the 32x32 tile
  const int arow = tid >> 2;          // 0..31
  const int acol = (tid & 3) * 8;     // halves: 0,8,16,24
  unsigned long long ga =
      (unsigned long long)(uintptr_t)(A + (size_t)(mb + arow) * lda + acol);
  unsigned lds0 = (unsigned)(uintptr_t)&sA[0][arow][acol];
  unsigned lds1 = (unsigned)(uintptr_t)&sA[1][arow][acol];

  // prologue: kick off first A tile
  asm volatile("global_load_async_to_lds_b128 %0, %1, off"
               :: "v"(lds0), "v"(ga) : "memory");

  v8f acc[2][4] = {};
  const int nsteps = K >> 5;

  for (int s = 0; s < nsteps; ++s) {
    // wait own async DMA, then block-wide barrier: tile (s&1) is ready and
    // the previous step's LDS readers are all retired.
    asm volatile("s_wait_asynccnt 0x0" ::: "memory");
    __syncthreads();
    if (s + 1 < nsteps) {  // overlap next tile's DMA with this tile's math
      unsigned long long gn = ga + (unsigned long long)(s + 1) * 64ull;
      unsigned ld = ((s + 1) & 1) ? lds1 : lds0;
      asm volatile("global_load_async_to_lds_b128 %0, %1, off"
                   :: "v"(ld), "v"(gn) : "memory");
    }
    const _Float16 (*cs)[32] = sA[s & 1];

    // A fragments per ISA 16-bit A 16x32 layout:
    //   halves 0..7  <- K = 8g + 0..7 ; halves 8..15 <- K = 16 + 8g + 0..7
    union { v16h v; v8h h[2]; } a0, a1;
    a0.h[0] = *(const v8h*)&cs[mh][8 * g];
    a0.h[1] = *(const v8h*)&cs[mh][16 + 8 * g];
    a1.h[0] = *(const v8h*)&cs[16 + mh][8 * g];
    a1.h[1] = *(const v8h*)&cs[16 + mh][16 + 8 * g];

    const int kb = s << 5;
#pragma unroll
    for (int t = 0; t < 4; ++t) {
      // B fragment: lane holds column nn, halves 0..7 -> K = 8g + h, 8..15 -> K = 16 + 8g + (h-8).
      const int nn = nb0 + t * 16 + mh;
      const _Float16* bp = Wt + (size_t)nn * K + kb + 8 * g;
      union { v16h v; v8h h[2]; } bf;
      bf.h[0] = *(const v8h*)bp;
      bf.h[1] = *(const v8h*)(bp + 16);
      __builtin_prefetch((const void*)(bp + 32), 0, 1);  // next K-step
      acc[0][t] = __builtin_amdgcn_wmma_f32_16x16x32_f16(
          false, a0.v, false, bf.v, (short)0, acc[0][t], false, false);
      acc[1][t] = __builtin_amdgcn_wmma_f32_16x16x32_f16(
          false, a1.v, false, bf.v, (short)0, acc[1][t], false, false);
    }
  }

  // epilogue: C element r of lane (g, mh) is row mb + mt*16 + r + 8g
#pragma unroll
  for (int mt = 0; mt < 2; ++mt) {
#pragma unroll
    for (int t = 0; t < 4; ++t) {
      const int col = nb0 + t * 16 + mh;
      if (col < N) {
        const float bv = bias ? bias[col] : 0.f;
#pragma unroll
        for (int r = 0; r < 8; ++r) {
          const int row = mb + mt * 16 + r + 8 * g;
          float x = acc[mt][t][r] + bv;
          if (ACTIV == 1) x = (x > 0.f) ? x : (expf(x) - 1.f);  // ELU
          if (C32) C32[(size_t)row * ldc32 + col] = x;
          if (C16) C16[(size_t)row * ldc16 + col] = (_Float16)x;
        }
      }
    }
  }
}

// ---------------------------- elementwise kernels --------------------------
__global__ void convw_kernel(const float* __restrict__ src,
                             _Float16* __restrict__ dst, int K, int N, int Npad,
                             int srcKN, int n) {
  int idx = blockIdx.x * blockDim.x + threadIdx.x;
  if (idx >= n) return;
  int nn = idx / K, k = idx % K;
  float v = 0.f;
  if (nn < N) v = srcKN ? src[(size_t)k * N + nn] : src[(size_t)nn * K + k];
  dst[idx] = (_Float16)v;
}

__global__ void zero_f32_kernel(float* p, int n) {
  int i = blockIdx.x * blockDim.x + threadIdx.x;
  if (i < n) p[i] = 0.f;
}
__global__ void zero_f16_kernel(_Float16* p, int n) {
  int i = blockIdx.x * blockDim.x + threadIdx.x;
  if (i < n) p[i] = (_Float16)0.f;
}

// dst[b*C+c] = src[(b*LL + t)*C + c], f32 -> f16
__global__ void slice_to_f16_kernel(const float* __restrict__ src,
                                    _Float16* __restrict__ dst, int t, int C,
                                    int n) {
  int idx = blockIdx.x * blockDim.x + threadIdx.x;
  if (idx >= n) return;
  int b = idx / C, c = idx % C;
  dst[idx] = (_Float16)src[((size_t)b * LL + t) * C + c];
}

// LayerNorm + ELU over H features; writes f16 (next GEMM) and/or f32.
__global__ __launch_bounds__(256) void ln_elu_kernel(
    const float* __restrict__ X, int ldx, const float* __restrict__ gamma,
    const float* __restrict__ beta, _Float16* __restrict__ O16, int ld16,
    float* __restrict__ O32, int ld32, int H) {
  const int b = blockIdx.x;
  const float* x = X + (size_t)b * ldx;
  float s = 0.f;
  for (int i = threadIdx.x; i < H; i += blockDim.x) s += x[i];
  const float mean = blockReduceSumAll(s) / (float)H;
  float vs = 0.f;
  for (int i = threadIdx.x; i < H; i += blockDim.x) {
    float d = x[i] - mean;
    vs += d * d;
  }
  const float var = blockReduceSumAll(vs) / (float)H;
  const float inv = rsqrtf(var + 1e-5f);
  for (int i = threadIdx.x; i < H; i += blockDim.x) {
    float v = gamma[i] * (x[i] - mean) * inv + beta[i];
    v = (v > 0.f) ? v : (expf(v) - 1.f);
    if (O16) O16[(size_t)b * ld16 + i] = (_Float16)v;
    if (O32) O32[(size_t)b * ld32 + i] = v;
  }
}

__global__ void add_x_kernel(const float* __restrict__ a,
                             const float* __restrict__ b,
                             _Float16* __restrict__ o, int n) {
  int i = blockIdx.x * blockDim.x + threadIdx.x;
  if (i < n) o[i] = (_Float16)(a[i] + b[i]);
}

// GRU cell combine (torch gate order r,z,n). Writes f32 h and f16 copies
// into post_in[:, :DET] and feat[:, :DET].
__global__ void gru_kernel(const float* __restrict__ gi,
                           const float* __restrict__ gh,
                           const float* __restrict__ hprev,
                           float* __restrict__ hnew,
                           _Float16* __restrict__ post_in,
                           _Float16* __restrict__ feat) {
  int idx = blockIdx.x * blockDim.x + threadIdx.x;
  if (idx >= BB * DET) return;
  int b = idx / DET, j = idx % DET;
  const float* gib = gi + (size_t)b * 3 * DET;
  const float* ghb = gh + (size_t)b * 3 * DET;
  float r = sigmoidf_(gib[j] + ghb[j]);
  float z = sigmoidf_(gib[DET + j] + ghb[DET + j]);
  float n = tanhf(gib[2 * DET + j] + r * ghb[2 * DET + j]);
  float h = (1.f - z) * n + z * hprev[idx];
  hnew[idx] = h;
  post_in[(size_t)b * (DET + HIDD) + j] = (_Float16)h;
  feat[(size_t)b * FEATD + j] = (_Float16)h;
}

// rsample + KL(post||prior) with free-bits; writes stoch into feat slice.
__global__ __launch_bounds__(256) void kl_stoch_kernel(
    const float* __restrict__ pri, const float* __restrict__ pos,
    const float* __restrict__ noise_t, _Float16* __restrict__ feat,
    float* __restrict__ accum) {
  int idx = blockIdx.x * blockDim.x + threadIdx.x;
  float kl = 0.f;
  if (idx < BB * STO) {
    int b = idx / STO, j = idx % STO;
    float pm = pri[(size_t)b * 2 * STO + j];
    float ps = softplusf_(pri[(size_t)b * 2 * STO + STO + j]) + 0.1f;
    float qm = pos[(size_t)b * 2 * STO + j];
    float qs = softplusf_(pos[(size_t)b * 2 * STO + STO + j]) + 0.1f;
    float st = qm + qs * noise_t[idx];
    feat[(size_t)b * FEATD + DET + j] = (_Float16)st;
    float dm = qm - pm;
    kl = logf(ps / qs) + (qs * qs + dm * dm) / (2.f * ps * ps) - 0.5f;
    kl = fmaxf(kl, 1.0f);  // FREE_BITS
  }
  float s = blockReduceSumAll(kl);
  if (threadIdx.x == 0) atomicAdd(accum + 0, s);
}

__global__ __launch_bounds__(256) void recon_kernel(
    const float* __restrict__ dec_out, const float* __restrict__ obs_seq,
    int t, float* __restrict__ accum) {
  int idx = blockIdx.x * blockDim.x + threadIdx.x;
  float d = 0.f;
  if (idx < BB * OBSD) {
    int b = idx >> 8, i = idx & 255;
    float df = dec_out[idx] - obs_seq[((size_t)b * LL + t) * OBSD + i];
    d = df * df;
  }
  float s = blockReduceSumAll(d);
  if (threadIdx.x == 0) atomicAdd(accum + 1, s);
}

// symlog two-hot cross-entropy reward loss; one block per batch row.
__global__ __launch_bounds__(256) void rew_loss_kernel(
    const float* __restrict__ logits, const float* __restrict__ rew_seq, int t,
    const float* __restrict__ bins, float* __restrict__ accum) {
  const int b = blockIdx.x, tid = threadIdx.x;
  const float* lg = logits + (size_t)b * 256;
  float r = rew_seq[(size_t)b * LL + t];
  float sgn = (r > 0.f) ? 1.f : ((r < 0.f) ? -1.f : 0.f);
  float x = sgn * log1pf(fabsf(r));  // symlog
  float cnt = blockReduceSumAll((tid < NB && bins[tid] <= x) ? 1.f : 0.f);
  int k = (int)cnt - 1;
  k = (k < 0) ? 0 : ((k > NB - 2) ? NB - 2 : k);
  float v = (tid < NB) ? lg[tid] : -3.0e38f;
  float mx = blockReduceMaxAll(v);
  float se = blockReduceSumAll((tid < NB) ? expf(lg[tid] - mx) : 0.f);
  if (tid == 0) {
    float lse = mx + logf(se);
    float lo = bins[k], hi = bins[k + 1];
    float wu = (x - lo) / (hi - lo);
    wu = (wu < 0.f) ? 0.f : ((wu > 1.f) ? 1.f : wu);
    float loss = -((1.f - wu) * (lg[k] - lse) + wu * (lg[k + 1] - lse));
    atomicAdd(accum + 2, loss);
  }
}

// continuation head: dot(hidden,W2)+b2 -> sigmoid -> BCE vs (1-done)
__global__ __launch_bounds__(256) void cont_head_kernel(
    const _Float16* __restrict__ hidden, const float* __restrict__ w2,
    const float* __restrict__ b2, const float* __restrict__ done_seq, int t,
    float* __restrict__ accum) {
  const int b = blockIdx.x, tid = threadIdx.x;
  float v = (float)hidden[(size_t)b * 256 + tid] * w2[tid];
  float s = blockReduceSumAll(v);
  if (tid == 0) {
    float cp = sigmoidf_(s + b2[0]);
    cp = fminf(fmaxf(cp, 1e-7f), 1.f - 1e-7f);
    float tgt = 1.f - done_seq[(size_t)b * LL + t];
    float loss = -(tgt * logf(cp) + (1.f - tgt) * logf(1.f - cp));
    atomicAdd(accum + 3, loss);
  }
}

__global__ void finalize_kernel(const float* __restrict__ accum,
                                float* __restrict__ out) {
  float kl = accum[0] / ((float)BB * STO * LL);
  float rc = accum[1] / ((float)BB * OBSD * LL);
  float rw = accum[2] / ((float)BB * LL);
  float ct = accum[3] / ((float)BB * LL);
  out[0] = kl + rc + rw + ct;
  out[1] = kl;
  out[2] = rc;
  out[3] = rw;
  out[4] = ct;
}

// ------------------------------- host side ---------------------------------
static inline void launch_gemm(hipStream_t s, int activ, const _Float16* A,
                               int lda, const _Float16* Wt, const float* bias,
                               float* C32, int ldc32, _Float16* C16, int ldc16,
                               int M, int N, int Npad, int K) {
  dim3 grid(Npad / 256, M / 32);
  if (activ)
    wmma_gemm<1><<<grid, 128, 0, s>>>(A, lda, Wt, bias, C32, ldc32, C16, ldc16,
                                      M, N, K);
  else
    wmma_gemm<0><<<grid, 128, 0, s>>>(A, lda, Wt, bias, C32, ldc32, C16, ldc16,
                                      M, N, K);
}

extern "C" void kernel_launch(void* const* d_in, const int* in_sizes, int n_in,
                              void* d_out, int out_size, void* d_ws,
                              size_t ws_size, hipStream_t stream) {
  (void)in_sizes; (void)n_in; (void)out_size; (void)ws_size;

  // inputs in setup_inputs() dict order (params dict flattened in order)
  const float* obs_seq  = (const float*)d_in[0];
  const float* act_seq  = (const float*)d_in[1];
  const float* rew_seq  = (const float*)d_in[2];
  const float* done_seq = (const float*)d_in[3];
  const float* noise    = (const float*)d_in[4];
  const float* bins     = (const float*)d_in[5];
  const float *enc_W1 = (const float*)d_in[6],  *enc_b1 = (const float*)d_in[7];
  const float *enc_g1 = (const float*)d_in[8],  *enc_be1 = (const float*)d_in[9];
  const float *enc_W2 = (const float*)d_in[10], *enc_b2 = (const float*)d_in[11];
  const float *enc_g2 = (const float*)d_in[12], *enc_be2 = (const float*)d_in[13];
  const float *dec_W1 = (const float*)d_in[14], *dec_b1 = (const float*)d_in[15];
  const float *dec_g1 = (const float*)d_in[16], *dec_be1 = (const float*)d_in[17];
  const float *dec_W2 = (const float*)d_in[18], *dec_b2 = (const float*)d_in[19];
  const float *dec_g2 = (const float*)d_in[20], *dec_be2 = (const float*)d_in[21];
  const float *dec_W3 = (const float*)d_in[22], *dec_b3 = (const float*)d_in[23];
  const float *gru_Wih = (const float*)d_in[24], *gru_Whh = (const float*)d_in[25];
  const float *gru_bih = (const float*)d_in[26], *gru_bhh = (const float*)d_in[27];
  const float *prior_W1 = (const float*)d_in[28], *prior_b1 = (const float*)d_in[29];
  const float *prior_W2 = (const float*)d_in[30], *prior_b2 = (const float*)d_in[31];
  const float *post_W1 = (const float*)d_in[32], *post_b1 = (const float*)d_in[33];
  const float *post_W2 = (const float*)d_in[34], *post_b2 = (const float*)d_in[35];
  const float *act_W = (const float*)d_in[36], *act_b = (const float*)d_in[37];
  const float *rew_W1 = (const float*)d_in[38], *rew_b1 = (const float*)d_in[39];
  const float *rew_W2 = (const float*)d_in[40], *rew_b2 = (const float*)d_in[41];
  const float *cont_W1 = (const float*)d_in[42], *cont_b1 = (const float*)d_in[43];
  const float *cont_W2 = (const float*)d_in[44], *cont_b2 = (const float*)d_in[45];

  // bump allocator over workspace
  char* base = (char*)d_ws;
  size_t off = 0;
  auto alloc = [&](size_t bytes) -> void* {
    off = (off + 255) & ~(size_t)255;
    void* p = base + off;
    off += bytes;
    return p;
  };
  auto a16 = [&](size_t n) { return (_Float16*)alloc(n * sizeof(_Float16)); };
  auto a32 = [&](size_t n) { return (float*)alloc(n * sizeof(float)); };

  // f16 weights, [Npad][K] (K contiguous)
  _Float16* wt_encW1 = a16(512 * 256);
  _Float16* wt_encW2 = a16(512 * 512);
  _Float16* wt_decW1 = a16(512 * 1088);
  _Float16* wt_decW2 = a16(512 * 512);
  _Float16* wt_decW3 = a16(256 * 512);
  _Float16* wt_Wih   = a16((size_t)3072 * 512);
  _Float16* wt_Whh   = a16((size_t)3072 * 1024);
  _Float16* wt_priW1 = a16(512 * 1024);
  _Float16* wt_priW2 = a16(256 * 512);
  _Float16* wt_posW1 = a16((size_t)512 * 1536);
  _Float16* wt_posW2 = a16(256 * 512);
  _Float16* wt_actW  = a16(512 * 32);
  _Float16* wt_rewW1 = a16(256 * 1088);
  _Float16* wt_rewW2 = a16(256 * 256);
  _Float16* wt_conW1 = a16(256 * 1088);

  // activations
  _Float16* obs16     = a16((size_t)BB * OBSD);
  _Float16* ac16      = a16((size_t)BB * ACTD);
  float*    f32a      = a32((size_t)BB * HIDD);
  float*    f32b      = a32((size_t)BB * HIDD);
  _Float16* h16a      = a16((size_t)BB * HIDD);
  _Float16* h16b      = a16((size_t)BB * HIDD);
  float*    obs_emb32 = a32((size_t)BB * HIDD);
  float*    act_emb32 = a32((size_t)BB * HIDD);
  _Float16* x16       = a16((size_t)BB * HIDD);
  float*    gi        = a32((size_t)BB * 3 * DET);
  float*    gh        = a32((size_t)BB * 3 * DET);
  float*    hA        = a32((size_t)BB * DET);
  float*    hB        = a32((size_t)BB * DET);
  _Float16* post_in   = a16((size_t)BB * (DET + HIDD));  // [h | obs_emb]
  _Float16* feat      = a16((size_t)BB * FEATD);         // [h | stoch]
  _Float16* ph16      = a16((size_t)BB * HIDD);
  _Float16* qh16      = a16((size_t)BB * HIDD);
  float*    pri       = a32((size_t)BB * 2 * STO);
  float*    pos       = a32((size_t)BB * 2 * STO);
  float*    dec_out   = a32((size_t)BB * OBSD);
  _Float16* rh16      = a16((size_t)BB * 256);
  float*    logits    = a32((size_t)BB * 256);
  _Float16* ch16      = a16((size_t)BB * 256);
  float*    accum     = a32(8);

  auto convw = [&](const float* src, _Float16* dst, int K, int N, int Npad,
                   int srcKN) {
    int n = Npad * K;
    convw_kernel<<<(n + 255) / 256, 256, 0, stream>>>(src, dst, K, N, Npad,
                                                      srcKN, n);
  };
  // srcKN=1: source stored [K][N]; srcKN=0: already [N][K]
  convw(enc_W1, wt_encW1, 256, 512, 512, 1);
  convw(enc_W2, wt_encW2, 512, 512, 512, 1);
  convw(dec_W1, wt_decW1, 1088, 512, 512, 1);
  convw(dec_W2, wt_decW2, 512, 512, 512, 1);
  convw(dec_W3, wt_decW3, 512, 256, 256, 1);
  convw(gru_Wih, wt_Wih, 512, 3072, 3072, 0);
  convw(gru_Whh, wt_Whh, 1024, 3072, 3072, 0);
  convw(prior_W1, wt_priW1, 1024, 512, 512, 1);
  convw(prior_W2, wt_priW2, 512, 128, 256, 1);
  convw(post_W1, wt_posW1, 1536, 512, 512, 1);
  convw(post_W2, wt_posW2, 512, 128, 256, 1);
  convw(act_W, wt_actW, 32, 512, 512, 1);
  convw(rew_W1, wt_rewW1, 1088, 256, 256, 1);
  convw(rew_W2, wt_rewW2, 256, 255, 256, 1);
  convw(cont_W1, wt_conW1, 1088, 256, 256, 1);

  zero_f32_kernel<<<1, 32, 0, stream>>>(accum, 8);
  {
    int n = BB * (DET + HIDD);
    zero_f16_kernel<<<(n + 255) / 256, 256, 0, stream>>>(post_in, n);
    zero_f32_kernel<<<(BB * DET + 255) / 256, 256, 0, stream>>>(hA, BB * DET);
  }

  float* hprev = hA;
  float* hnext = hB;
  const int LD_PI = DET + HIDD;  // 1536

  for (int t = 0; t < LL; ++t) {
    // slice obs/act step -> f16
    slice_to_f16_kernel<<<(BB * OBSD + 255) / 256, 256, 0, stream>>>(
        obs_seq, obs16, t, OBSD, BB * OBSD);
    slice_to_f16_kernel<<<(BB * ACTD + 255) / 256, 256, 0, stream>>>(
        act_seq, ac16, t, ACTD, BB * ACTD);

    // encoder: GEMM -> LN+ELU -> GEMM -> LN+ELU
    launch_gemm(stream, 0, obs16, OBSD, wt_encW1, enc_b1, f32a, HIDD, nullptr,
                0, BB, HIDD, 512, OBSD);
    ln_elu_kernel<<<BB, 256, 0, stream>>>(f32a, HIDD, enc_g1, enc_be1, h16a,
                                          HIDD, nullptr, 0, HIDD);
    launch_gemm(stream, 0, h16a, HIDD, wt_encW2, enc_b2, f32b, HIDD, nullptr,
                0, BB, HIDD, 512, HIDD);
    ln_elu_kernel<<<BB, 256, 0, stream>>>(f32b, HIDD, enc_g2, enc_be2,
                                          post_in + DET, LD_PI, obs_emb32,
                                          HIDD, HIDD);

    // action embedding: ELU(act @ W + b)
    launch_gemm(stream, 1, ac16, ACTD, wt_actW, act_b, act_emb32, HIDD,
                nullptr, 0, BB, HIDD, 512, ACTD);
    add_x_kernel<<<(BB * HIDD + 255) / 256, 256, 0, stream>>>(
        obs_emb32, act_emb32, x16, BB * HIDD);

    // GRU: gi = x@Wih.T+bih ; gh = h@Whh.T+bhh (h f16 lives in post_in[:, :DET])
    launch_gemm(stream, 0, x16, HIDD, wt_Wih, gru_bih, gi, 3 * DET, nullptr, 0,
                BB, 3 * DET, 3 * DET, HIDD);
    launch_gemm(stream, 0, post_in, LD_PI, wt_Whh, gru_bhh, gh, 3 * DET,
                nullptr, 0, BB, 3 * DET, 3 * DET, DET);
    gru_kernel<<<(BB * DET + 255) / 256, 256, 0, stream>>>(gi, gh, hprev,
                                                           hnext, post_in,
                                                           feat);
    { float* tmp = hprev; hprev = hnext; hnext = tmp; }

    // prior head
    launch_gemm(stream, 1, post_in, LD_PI, wt_priW1, prior_b1, nullptr, 0,
                ph16, HIDD, BB, HIDD, 512, DET);
    launch_gemm(stream, 0, ph16, HIDD, wt_priW2, prior_b2, pri, 2 * STO,
                nullptr, 0, BB, 2 * STO, 256, HIDD);
    // posterior head on [deter2 | obs_emb]
    launch_gemm(stream, 1, post_in, LD_PI, wt_posW1, post_b1, nullptr, 0, qh16,
                HIDD, BB, HIDD, 512, DET + HIDD);
    launch_gemm(stream, 0, qh16, HIDD, wt_posW2, post_b2, pos, 2 * STO,
                nullptr, 0, BB, 2 * STO, 256, HIDD);
    kl_stoch_kernel<<<(BB * STO + 255) / 256, 256, 0, stream>>>(
        pri, pos, noise + (size_t)t * BB * STO, feat, accum);

    // decoder
    launch_gemm(stream, 0, feat, FEATD, wt_decW1, dec_b1, f32a, HIDD, nullptr,
                0, BB, HIDD, 512, FEATD);
    ln_elu_kernel<<<BB, 256, 0, stream>>>(f32a, HIDD, dec_g1, dec_be1, h16a,
                                          HIDD, nullptr, 0, HIDD);
    launch_gemm(stream, 0, h16a, HIDD, wt_decW2, dec_b2, f32b, HIDD, nullptr,
                0, BB, HIDD, 512, HIDD);
    ln_elu_kernel<<<BB, 256, 0, stream>>>(f32b, HIDD, dec_g2, dec_be2, h16b,
                                          HIDD, nullptr, 0, HIDD);
    launch_gemm(stream, 0, h16b, HIDD, wt_decW3, dec_b3, dec_out, OBSD,
                nullptr, 0, BB, OBSD, 256, HIDD);
    recon_kernel<<<(BB * OBSD + 255) / 256, 256, 0, stream>>>(dec_out, obs_seq,
                                                              t, accum);

    // reward head
    launch_gemm(stream, 1, feat, FEATD, wt_rewW1, rew_b1, nullptr, 0, rh16,
                256, BB, 256, 256, FEATD);
    launch_gemm(stream, 0, rh16, 256, wt_rewW2, rew_b2, logits, 256, nullptr,
                0, BB, NB, 256, 256);
    rew_loss_kernel<<<BB, 256, 0, stream>>>(logits, rew_seq, t, bins, accum);

    // continuation head
    launch_gemm(stream, 1, feat, FEATD, wt_conW1, cont_b1, nullptr, 0, ch16,
                256, BB, 256, 256, FEATD);
    cont_head_kernel<<<BB, 256, 0, stream>>>(ch16, cont_W2, cont_b2, done_seq,
                                             t, accum);
  }

  finalize_kernel<<<1, 1, 0, stream>>>(accum, (float*)d_out);
}
